// AttentionBlock_10299331576132
// MI455X (gfx1250) — compile-verified, matrix operand order checked by reading, not run on hardware
//
#include <hip/hip_runtime.h>
#include <math.h>
#include <stdint.h>

// ---------------- CDNA5 WMMA types ----------------
typedef __bf16 v16bf __attribute__((ext_vector_type(16)));
typedef float  v8f   __attribute__((ext_vector_type(8)));
typedef unsigned int u32x4 __attribute__((ext_vector_type(4)));
typedef unsigned int u32x8 __attribute__((ext_vector_type(8)));

#define WMMA_BF16(a, b, c) \
  __builtin_amdgcn_wmma_f32_16x16x32_bf16(false, (a), false, (b), (short)0, (c), false, false)

// Problem constants (fixed by the reference)
enum { B_ = 2, S_ = 2048, H_ = 16, D_ = 64, E_ = 1024, FC_ = 4096, R_ = B_ * S_ };

#define LDA 72  // padded LDS row stride (bf16): 64 data + 8 pad = 144B rows, 16B aligned

// Async global->LDS copy of 16 bytes (per lane). LDS dest = low 32 bits of generic pointer.
__device__ __forceinline__ void async_cp16(const void* lds, const void* gptr) {
  unsigned l = (unsigned)(uintptr_t)lds;
  unsigned long long g = (unsigned long long)(uintptr_t)gptr;
  asm volatile("global_load_async_to_lds_b128 %0, %1, off" ::"v"(l), "v"(g) : "memory");
}
__device__ __forceinline__ void async_wait0() {
  asm volatile("s_wait_asynccnt 0x0" ::: "memory");
}

// Tensor Data Mover: 2D tile (64 bf16 per row x `rows` rows, global row stride
// `stride_elems`) -> LDS at byte offset `ldsoff`, with 16B pad after each 128B row
// (produces the LDA=72 padded layout). D# per ISA ch.8; issued once per wave.
__device__ __forceinline__ void tdm_load_tile64(unsigned ldsoff, const void* gsrc,
                                                unsigned rows, unsigned stride_elems) {
  unsigned long long ga = (unsigned long long)(uintptr_t)gsrc;
  u32x4 g0;
  g0[0] = 1u;                                   // count=1 (valid user descriptor)
  g0[1] = ldsoff;                               // lds_addr (bytes)
  g0[2] = (unsigned)ga;                         // global_addr[31:0]
  g0[3] = (unsigned)(ga >> 32) | 0x80000000u;   // global_addr[56:32] | type=2<<30
  u32x8 g1;
  g1[0] = (1u << 16)      // data_size = 2 bytes
        | (1u << 20)      // pad_enable
        | (4u << 22)      // pad_interval: 32 DWORDs (=128B row)
        | (3u << 25);     // pad_amount: 4 DWORDs (=16B pad)
  g1[1] = 0u;                                   // atomic barrier addr / tensor_dim0 lo16=0
  g1[2] = 0x4000u;                              // tensor_dim0 = 1<<30 (no OOB clipping)
  g1[3] = 0x4000u | (64u << 16);                // tensor_dim1 = 1<<30; tile_dim0 = 64
  g1[4] = rows;                                 // tile_dim1 = rows; tile_dim2 = 0
  g1[5] = stride_elems;                         // tensor_dim0_stride[31:0]
  g1[6] = 0u;
  g1[7] = 0u;
  asm volatile("tensor_load_to_lds %0, %1" ::"s"(g0), "s"(g1) : "memory");
}

// ---------------- fragment loader (ISA 7.12.2 layout) ----------------
__device__ __forceinline__ v16bf frag_a(const __bf16* p, int ld, int half, int m) {
  v16bf f;
  const __bf16* r = p + m * ld;
#pragma unroll
  for (int j = 0; j < 8; ++j) f[j] = r[half * 8 + j];
#pragma unroll
  for (int j = 0; j < 8; ++j) f[8 + j] = r[16 + half * 8 + j];
  return f;
}

__device__ __forceinline__ float rowmax16(float v) {
  v = fmaxf(v, __shfl_xor(v, 1, 32));
  v = fmaxf(v, __shfl_xor(v, 2, 32));
  v = fmaxf(v, __shfl_xor(v, 4, 32));
  v = fmaxf(v, __shfl_xor(v, 8, 32));
  return v;
}
__device__ __forceinline__ float rowsum16(float v) {
  v += __shfl_xor(v, 1, 32);
  v += __shfl_xor(v, 2, 32);
  v += __shfl_xor(v, 4, 32);
  v += __shfl_xor(v, 8, 32);
  return v;
}

// ---------------- LayerNorm kernels ----------------
__global__ void ln_f32_to_bf16(const float* __restrict__ in, const float* __restrict__ w,
                               const float* __restrict__ bb, __bf16* __restrict__ out, int cols) {
  int row = blockIdx.x;
  const float* p = in + (size_t)row * cols;
  float s = 0.f, s2 = 0.f;
  for (int c = threadIdx.x; c < cols; c += blockDim.x) { float v = p[c]; s += v; s2 += v * v; }
  __shared__ float sh[2][8];
#pragma unroll
  for (int o = 16; o > 0; o >>= 1) { s += __shfl_down(s, o, 32); s2 += __shfl_down(s2, o, 32); }
  if ((threadIdx.x & 31) == 0) { sh[0][threadIdx.x >> 5] = s; sh[1][threadIdx.x >> 5] = s2; }
  __syncthreads();
  float ts = 0.f, ts2 = 0.f;
#pragma unroll
  for (int i = 0; i < 8; ++i) { ts += sh[0][i]; ts2 += sh[1][i]; }
  float mu = ts / cols;
  float rstd = rsqrtf(ts2 / cols - mu * mu + 1e-5f);
  __bf16* o = out + (size_t)row * cols;
  for (int c = threadIdx.x; c < cols; c += blockDim.x)
    o[c] = (__bf16)((p[c] - mu) * rstd * w[c] + bb[c]);
}

__global__ void ln_add_res(const float* __restrict__ a, const float* __restrict__ w,
                           const float* __restrict__ bb, const float* __restrict__ resid,
                           float* __restrict__ out, int cols) {
  int row = blockIdx.x;
  const float* p = a + (size_t)row * cols;
  float s = 0.f, s2 = 0.f;
  for (int c = threadIdx.x; c < cols; c += blockDim.x) { float v = p[c]; s += v; s2 += v * v; }
  __shared__ float sh[2][8];
#pragma unroll
  for (int o = 16; o > 0; o >>= 1) { s += __shfl_down(s, o, 32); s2 += __shfl_down(s2, o, 32); }
  if ((threadIdx.x & 31) == 0) { sh[0][threadIdx.x >> 5] = s; sh[1][threadIdx.x >> 5] = s2; }
  __syncthreads();
  float ts = 0.f, ts2 = 0.f;
#pragma unroll
  for (int i = 0; i < 8; ++i) { ts += sh[0][i]; ts2 += sh[1][i]; }
  float mu = ts / cols;
  float rstd = rsqrtf(ts2 / cols - mu * mu + 1e-5f);
  float* o = out + (size_t)row * cols;
  const float* rr = resid + (size_t)row * cols;
  for (int c = threadIdx.x; c < cols; c += blockDim.x)
    o[c] = (p[c] - mu) * rstd * w[c] + bb[c] + rr[c];
}

__global__ void ln_bf16_inplace(__bf16* __restrict__ g, const float* __restrict__ w,
                                const float* __restrict__ bb, int cols) {
  int row = blockIdx.x;
  __bf16* p = g + (size_t)row * cols;
  float s = 0.f, s2 = 0.f;
  for (int c = threadIdx.x; c < cols; c += blockDim.x) { float v = (float)p[c]; s += v; s2 += v * v; }
  __shared__ float sh[2][8];
#pragma unroll
  for (int o = 16; o > 0; o >>= 1) { s += __shfl_down(s, o, 32); s2 += __shfl_down(s2, o, 32); }
  if ((threadIdx.x & 31) == 0) { sh[0][threadIdx.x >> 5] = s; sh[1][threadIdx.x >> 5] = s2; }
  __syncthreads();
  float ts = 0.f, ts2 = 0.f;
#pragma unroll
  for (int i = 0; i < 8; ++i) { ts += sh[0][i]; ts2 += sh[1][i]; }
  float mu = ts / cols;
  float rstd = rsqrtf(ts2 / cols - mu * mu + 1e-5f);
  for (int c = threadIdx.x; c < cols; c += blockDim.x)
    p[c] = (__bf16)(((float)p[c] - mu) * rstd * w[c] + bb[c]);
}

// ---------------- tiled WMMA GEMM: C[MxN] = A_bf16[MxK] @ B_f32[KxN] ----------------
// 256 threads / 8 waves (4x2); tile 128x64, BK=64; double-buffered LDS.
// A tile staged by the TDM (tensor_load_to_lds); B tile converted f32->bf16 and
// stored transposed [n][k] so both operands load as contiguous ds_load_b128 frags.
template <int EPI>
__global__ __launch_bounds__(256) void gemm128(const __bf16* __restrict__ A,
                                               const float* __restrict__ Bw, int M, int N, int K,
                                               float* __restrict__ Cf, __bf16* __restrict__ Cbf,
                                               const float* __restrict__ bias,
                                               const float* __restrict__ resid,
                                               const float* __restrict__ lam) {
  __shared__ __align__(16) __bf16 Asm[2][128 * LDA];
  __shared__ __align__(16) __bf16 Bsm[2][64 * LDA];  // transposed: [n][k]
  const int tid = threadIdx.x;
  const int wave = tid >> 5, lane = tid & 31;
  const int wr = wave >> 1, wc = wave & 1;
  const int half = lane >> 4, ln16 = lane & 15;
  const int m0 = blockIdx.y * 128, n0 = blockIdx.x * 64;
  const int bn = tid & 63, bkg = tid >> 6;  // B staging: column, k-group of 16
  v8f acc[2][2] = {};
  const int nk = K >> 6;

  // ---- staging helper (TDM for A, manual transpose+convert for B) ----
  auto stage = [&](int i) {
    const int k0 = i << 6;
    __bf16* As = Asm[i & 1];
    __bf16* Bs = Bsm[i & 1];
    if (wave == 0)
      tdm_load_tile64((unsigned)(uintptr_t)&As[0],
                      (const unsigned short*)A + (size_t)m0 * K + k0, 128u, (unsigned)K);
    unsigned short tmp[16];
#pragma unroll
    for (int t = 0; t < 16; ++t) {
      float f = Bw[(size_t)(k0 + bkg * 16 + t) * N + n0 + bn];
      __bf16 h = (__bf16)f;
      tmp[t] = *(unsigned short*)&h;
    }
#pragma unroll
    for (int t = 0; t < 4; ++t)
      *(ushort4*)&Bs[bn * LDA + bkg * 16 + t * 4] = *(ushort4*)&tmp[t * 4];
    if (k0 + 64 < K)
      __builtin_prefetch(Bw + (size_t)(k0 + 64 + bkg * 16) * N + n0 + bn, 0, 1);
  };

  stage(0);
  __builtin_amdgcn_s_wait_tensorcnt(0);
  __syncthreads();

  for (int i = 0; i < nk; ++i) {
    if (i + 1 < nk) stage(i + 1);  // overlap next-tile DMA with this tile's WMMAs
    const __bf16* As = Asm[i & 1];
    const __bf16* Bs = Bsm[i & 1];
#pragma unroll
    for (int kc = 0; kc < 2; ++kc) {
      v16bf a0 = frag_a(&As[(wr * 32) * LDA + kc * 32], LDA, half, ln16);
      v16bf a1 = frag_a(&As[(wr * 32 + 16) * LDA + kc * 32], LDA, half, ln16);
      v16bf b0 = frag_a(&Bs[(wc * 32) * LDA + kc * 32], LDA, half, ln16);
      v16bf b1 = frag_a(&Bs[(wc * 32 + 16) * LDA + kc * 32], LDA, half, ln16);
      acc[0][0] = WMMA_BF16(a0, b0, acc[0][0]);
      acc[0][1] = WMMA_BF16(a0, b1, acc[0][1]);
      acc[1][0] = WMMA_BF16(a1, b0, acc[1][0]);
      acc[1][1] = WMMA_BF16(a1, b1, acc[1][1]);
    }
    __builtin_amdgcn_s_wait_tensorcnt(0);
    __syncthreads();
  }

#pragma unroll
  for (int i = 0; i < 2; ++i)
#pragma unroll
    for (int j = 0; j < 2; ++j) {
      int colg = n0 + wc * 32 + j * 16 + ln16;
#pragma unroll
      for (int r = 0; r < 8; ++r) {
        int rowg = m0 + wr * 32 + i * 16 + r + 8 * half;
        float v = acc[i][j][r];
        size_t idx = (size_t)rowg * N + colg;
        if (EPI == 0) {
          Cbf[idx] = (__bf16)v;
        } else if (EPI == 1) {
          Cf[idx] = v;
        } else if (EPI == 2) {
          v += bias[colg];
          float g = 0.5f * v * (1.f + erff(v * 0.70710678118654752f));
          Cbf[idx] = (__bf16)g;
        } else {
          v += bias[colg];
          Cf[idx] = v + lam[colg] * resid[idx];
        }
      }
    }
}

// ---------------- flash attention: per (b,h) head, 64 q rows per block ----------------
__global__ __launch_bounds__(128) void attn_kernel(const __bf16* __restrict__ Q,
                                                   const __bf16* __restrict__ Kb,
                                                   const __bf16* __restrict__ V,
                                                   const unsigned char* __restrict__ mask,
                                                   const float* __restrict__ gamma,
                                                   __bf16* __restrict__ O) {
  __shared__ __align__(16) __bf16 Qs[64 * LDA];         // [q row][d]
  __shared__ __align__(16) __bf16 Ksm[2][64 * LDA];     // [key][d], double buffered
  __shared__ __align__(16) __bf16 Vsm[2][64 * LDA];     // transposed [d][key], double buffered
  __shared__ __align__(16) __bf16 Ps[4][16 * LDA];      // per-wave P scratch [m][key]

  const int tid = threadIdx.x, wave = tid >> 5, lane = tid & 31;
  const int half = lane >> 4, ln16 = lane & 15;
  const int bh = blockIdx.y, b = bh >> 4, h = bh & 15;
  const int s0 = blockIdx.x * 64;
  const size_t base = ((size_t)b * S_ * H_ + h) * D_;  // (b,0,h,0); row stride H_*D_
  const int vd = tid & 63, vkg = tid >> 6;             // V transpose staging coords

  auto stageKV = [&](int it) {
    const int kt = it << 6;
    __bf16* Ks = Ksm[it & 1];
    __bf16* Vs = Vsm[it & 1];
#pragma unroll
    for (int c = 0; c < 4; ++c) {  // K tile: async DMA straight copy into padded rows
      int e = (c * 128 + tid) * 8;
      int r = e >> 6, col = e & 63;
      async_cp16(&Ks[r * LDA + col],
                 (const unsigned short*)Kb + base + (size_t)(kt + r) * (H_ * D_) + col);
    }
    unsigned short vt[32];  // V tile: transpose during staging -> Vs[d][key]
#pragma unroll
    for (int t = 0; t < 32; ++t)
      vt[t] = ((const unsigned short*)V)[base + (size_t)(kt + vkg * 32 + t) * (H_ * D_) + vd];
#pragma unroll
    for (int t = 0; t < 8; ++t)
      *(ushort4*)&Vs[vd * LDA + vkg * 32 + t * 4] = *(ushort4*)&vt[t * 4];
  };

  // prologue: Q tile async + first K/V tile
#pragma unroll
  for (int c = 0; c < 4; ++c) {
    int e = (c * 128 + tid) * 8;
    int r = e >> 6, col = e & 63;
    async_cp16(&Qs[r * LDA + col],
               (const unsigned short*)Q + base + (size_t)(s0 + r) * (H_ * D_) + col);
  }
  stageKV(0);
  async_wait0();
  __syncthreads();
  const v16bf qf0 = frag_a(&Qs[(wave * 16) * LDA], LDA, half, ln16);
  const v16bf qf1 = frag_a(&Qs[(wave * 16) * LDA + 32], LDA, half, ln16);

  float mrow[8], lrow[8];
  v8f Oa[4] = {};
#pragma unroll
  for (int r = 0; r < 8; ++r) { mrow[r] = -1e30f; lrow[r] = 0.f; }

  const int nkt = S_ >> 6;
  for (int it = 0; it < nkt; ++it) {
    if (it + 1 < nkt) stageKV(it + 1);  // overlap next K/V fetch with this tile's math
    const __bf16* Ks = Ksm[it & 1];
    const __bf16* Vs = Vsm[it & 1];
    const int kt = it << 6;

    // scores: S = (Q @ K^T) / sqrt(D)
    float sc[4][8];
#pragma unroll
    for (int nt = 0; nt < 4; ++nt) {
      v16bf kf0 = frag_a(&Ks[(nt * 16) * LDA], LDA, half, ln16);
      v16bf kf1 = frag_a(&Ks[(nt * 16) * LDA + 32], LDA, half, ln16);
      v8f a = {};
      a = WMMA_BF16(qf0, kf0, a);
      a = WMMA_BF16(qf1, kf1, a);
      bool km = mask[b * S_ + kt + nt * 16 + ln16] != 0;
#pragma unroll
      for (int r = 0; r < 8; ++r) sc[nt][r] = km ? -1e30f : a[r] * 0.125f;
    }

    // online softmax (rows at vgpr r + 8*half; columns across 16 lanes)
    float corr[8];
#pragma unroll
    for (int r = 0; r < 8; ++r) {
      float mt = fmaxf(fmaxf(sc[0][r], sc[1][r]), fmaxf(sc[2][r], sc[3][r]));
      mt = rowmax16(mt);
      float mn = fmaxf(mrow[r], mt);
      corr[r] = __expf(mrow[r] - mn);
      float ps = 0.f;
#pragma unroll
      for (int nt = 0; nt < 4; ++nt) {
        float p = __expf(sc[nt][r] - mn);
        sc[nt][r] = p;
        ps += p;
      }
      ps = rowsum16(ps);
      lrow[r] = lrow[r] * corr[r] + ps;
      mrow[r] = mn;
    }

    // P (16x64 bf16) to per-wave LDS, reload as A-frags for P @ V
    __bf16* Pw = &Ps[wave][0];
#pragma unroll
    for (int nt = 0; nt < 4; ++nt)
#pragma unroll
      for (int r = 0; r < 8; ++r) Pw[(r + 8 * half) * LDA + nt * 16 + ln16] = (__bf16)sc[nt][r];
    v16bf pf0 = frag_a(Pw, LDA, half, ln16);
    v16bf pf1 = frag_a(Pw + 32, LDA, half, ln16);
#pragma unroll
    for (int dt = 0; dt < 4; ++dt) {
      v16bf vf0 = frag_a(&Vs[(dt * 16) * LDA], LDA, half, ln16);       // keys 0..31
      v16bf vf1 = frag_a(&Vs[(dt * 16) * LDA + 32], LDA, half, ln16);  // keys 32..63
      v8f o = Oa[dt];
#pragma unroll
      for (int r = 0; r < 8; ++r) o[r] *= corr[r];
      o = WMMA_BF16(pf0, vf0, o);
      o = WMMA_BF16(pf1, vf1, o);
      Oa[dt] = o;
    }
    async_wait0();
    __syncthreads();
  }

  const float g = gamma[h];
#pragma unroll
  for (int dt = 0; dt < 4; ++dt)
#pragma unroll
    for (int r = 0; r < 8; ++r) {
      int sg = s0 + wave * 16 + r + 8 * half;
      float val = Oa[dt][r] / lrow[r] * g;
      O[base + (size_t)sg * (H_ * D_) + dt * 16 + ln16] = (__bf16)val;
    }
}

// ---------------- launcher ----------------
extern "C" void kernel_launch(void* const* d_in, const int* in_sizes, int n_in, void* d_out,
                              int out_size, void* d_ws, size_t ws_size, hipStream_t stream) {
  (void)in_sizes; (void)n_in; (void)out_size; (void)ws_size;
  const float* x    = (const float*)d_in[0];
  const unsigned char* kpm = (const unsigned char*)d_in[1];
  const float* Wq   = (const float*)d_in[2];
  const float* Wk   = (const float*)d_in[3];
  const float* Wv   = (const float*)d_in[4];
  const float* Wo   = (const float*)d_in[5];
  const float* gamma= (const float*)d_in[6];
  const float* ln1w = (const float*)d_in[7];
  const float* ln1b = (const float*)d_in[8];
  const float* ln2w = (const float*)d_in[9];
  const float* ln2b = (const float*)d_in[10];
  const float* ln3w = (const float*)d_in[11];
  const float* ln3b = (const float*)d_in[12];
  const float* ln4w = (const float*)d_in[13];
  const float* ln4b = (const float*)d_in[14];
  const float* W1   = (const float*)d_in[15];
  const float* b1   = (const float*)d_in[16];
  const float* W2   = (const float*)d_in[17];
  const float* b2   = (const float*)d_in[18];
  const float* lam  = (const float*)d_in[19];

  char* ws = (char*)d_ws;
  const size_t MB = (size_t)1 << 20;
  __bf16* hbf  = (__bf16*)(ws + 0);        // R x E bf16 (LN1 & LN3 output)
  __bf16* qbf  = (__bf16*)(ws + 8 * MB);
  __bf16* kbf  = (__bf16*)(ws + 16 * MB);
  __bf16* vbf  = (__bf16*)(ws + 24 * MB);
  __bf16* wsbf = (__bf16*)(ws + 32 * MB);  // attention out * gamma
  float*  af32 = (float*)(ws + 40 * MB);   // R x E f32 (pre-LN2)
  float*  x2   = (float*)(ws + 56 * MB);   // R x E f32 (post-attn residual)
  __bf16* gbf  = (__bf16*)(ws + 72 * MB);  // R x FC bf16 (GELU / LN4)
  float*  out  = (float*)d_out;

  // 1) h = LN1(x) -> bf16
  ln_f32_to_bf16<<<R_, 256, 0, stream>>>(x, ln1w, ln1b, hbf, E_);
  // 2) QKV projections
  dim3 gE(E_ / 64, R_ / 128);
  gemm128<0><<<gE, 256, 0, stream>>>(hbf, Wq, R_, E_, E_, nullptr, qbf, nullptr, nullptr, nullptr);
  gemm128<0><<<gE, 256, 0, stream>>>(hbf, Wk, R_, E_, E_, nullptr, kbf, nullptr, nullptr, nullptr);
  gemm128<0><<<gE, 256, 0, stream>>>(hbf, Wv, R_, E_, E_, nullptr, vbf, nullptr, nullptr, nullptr);
  // 3) flash attention (+ per-head gamma)
  attn_kernel<<<dim3(S_ / 64, B_ * H_), 128, 0, stream>>>(qbf, kbf, vbf, kpm, gamma, wsbf);
  // 4) a = ws @ Wo
  gemm128<1><<<gE, 256, 0, stream>>>(wsbf, Wo, R_, E_, E_, af32, nullptr, nullptr, nullptr, nullptr);
  // 5) x2 = LN2(a) + x
  ln_add_res<<<R_, 256, 0, stream>>>(af32, ln2w, ln2b, x, x2, E_);
  // 6) h = LN3(x2) -> bf16
  ln_f32_to_bf16<<<R_, 256, 0, stream>>>(x2, ln3w, ln3b, hbf, E_);
  // 7) g = gelu(h @ W1 + b1) -> bf16
  gemm128<2><<<dim3(FC_ / 64, R_ / 128), 256, 0, stream>>>(hbf, W1, R_, FC_, E_, nullptr, gbf, b1,
                                                           nullptr, nullptr);
  // 8) g = LN4(g) in place
  ln_bf16_inplace<<<R_, 256, 0, stream>>>(gbf, ln4w, ln4b, FC_);
  // 9) out = g @ W2 + b2 + lam * x2
  gemm128<3><<<gE, 256, 0, stream>>>(gbf, W2, R_, E_, FC_, out, nullptr, b2, x2, lam);
}
